// TransformerBlock_76098230550591
// MI455X (gfx1250) — compile-verified, matrix operand order checked by reading, not run on hardware
//
#include <hip/hip_runtime.h>
#include <hip/hip_bf16.h>

// MI455X / gfx1250, wave32. All matrix math via v_wmma_f32_16x16x32_f16.
// Fragment loads are vectorized: CDNA5 16-bit A-fragment per lane is two
// contiguous 16B runs (K = 8*hs..+7 and 16+8*hs..+7), so A tiles (row-major
// in K) and B tiles (staged transposed, [n][k]) load as 2x ds_load_b128.

typedef __attribute__((ext_vector_type(16))) _Float16 v16h;
typedef __attribute__((ext_vector_type(8)))  float    v8f;

__device__ __forceinline__ v8f vzero8() {
  v8f z = {0.f, 0.f, 0.f, 0.f, 0.f, 0.f, 0.f, 0.f};
  return z;
}

__device__ __forceinline__ int kmap(int j, int hs) {
  return ((j < 8) ? j : (j + 8)) + (hs << 3);
}

// Vectorized fragment load: works for A tiles stored [m][k] (ld = leading dim,
// multiple of 8) and for B tiles stored transposed [n][k]. 2x ds_load_b128.
__device__ __forceinline__ v16h ldsFrag(const _Float16* p, int ld) {
  const int lane = threadIdx.x & 31;
  const int r = lane & 15, hs = lane >> 4;
  const _Float16* q0 = p + r * ld + 8 * hs;
  union {
    int4 i[2];
    v16h h;
  } u;
  u.i[0] = *reinterpret_cast<const int4*>(q0);
  u.i[1] = *reinterpret_cast<const int4*>(q0 + 16);
  return u.h;
}

// Scalar A fragment from transposed storage S[k][m] (small kernels only)
__device__ __forceinline__ v16h ldsAT(const _Float16* p, int ld) {
  const int lane = threadIdx.x & 31;
  const int r = lane & 15, hs = lane >> 4;
  v16h a;
#pragma unroll
  for (int j = 0; j < 16; ++j) a[j] = p[kmap(j, hs) * ld + r];
  return a;
}

// Scalar B fragment from natural storage B[k][n] (small kernels only)
__device__ __forceinline__ v16h ldsB(const _Float16* p, int ld) {
  const int lane = threadIdx.x & 31;
  const int c = lane & 15, hs = lane >> 4;
  v16h b;
#pragma unroll
  for (int j = 0; j < 16; ++j) b[j] = p[kmap(j, hs) * ld + c];
  return b;
}

__device__ __forceinline__ v8f wmma_f16(v16h a, v16h b, v8f c) {
  return __builtin_amdgcn_wmma_f32_16x16x32_f16(false, a, false, b, (short)0, c,
                                                false, false);
}

// ---------------- weight / input format conversions ----------------

// w_qkvv is [o][c]; B is used transposed ([n=o][k=c]) -> straight f16 copy.
__global__ __launch_bounds__(256) void k_cvt_wq(const float* __restrict__ w,
                                                _Float16* __restrict__ wqt) {
  size_t i = (size_t)blockIdx.x * 256 + threadIdx.x;
  wqt[i] = (_Float16)w[i];
}

__global__ __launch_bounds__(256) void k_cvt_ef(const float* __restrict__ ef,
                                                _Float16* __restrict__ eft) {
  int k = blockIdx.x;  // 0..63
  int n = blockIdx.y * 256 + threadIdx.x;
  eft[(size_t)k * 4096 + n] = (_Float16)ef[(size_t)n * 64 + k];
}

// conv w OIHW -> [tap][co][ci] (transposed-B layout per tap)
__global__ __launch_bounds__(256) void k_cvt_w9(const float* __restrict__ w,
                                                _Float16* __restrict__ w9t) {
  int tap = blockIdx.x;  // 0..8
  int co = blockIdx.y;   // 0..255
  int ci = threadIdx.x;
  w9t[((size_t)tap * 256 + co) * 256 + ci] =
      (_Float16)w[((size_t)co * 256 + ci) * 9 + tap];
}

// conv8 1x1 OI -> [co][ci] straight copy (transposed-B layout)
__global__ __launch_bounds__(256) void k_cvt_w8(const float* __restrict__ w,
                                                _Float16* __restrict__ w8t) {
  size_t i = (size_t)blockIdx.x * 256 + threadIdx.x;
  w8t[i] = (_Float16)w[i];
}

// ---------------- LayerNorm (NCHW -> f16 [BN][C]) ----------------

__global__ __launch_bounds__(256) void k_layernorm(const float* __restrict__ x,
                                                   const float* __restrict__ lnw,
                                                   const float* __restrict__ lnb,
                                                   _Float16* __restrict__ hout) {
  const int tid = threadIdx.x;
  const int bn0 = blockIdx.x * 256;
  const int bn = bn0 + tid;
  const int b = bn >> 12;
  const int n = bn & 4095;
  const float* xb = x + ((size_t)b << 20);
  float s = 0.f, sq = 0.f;
  for (int c = 0; c < 256; ++c) {
    float v = xb[((size_t)c << 12) + n];
    s += v;
    sq += v * v;
  }
  const float mean = s * (1.f / 256.f);
  const float var = sq * (1.f / 256.f) - mean * mean;
  const float rstd = rsqrtf(var + 1e-5f);

  __shared__ alignas(16) _Float16 tile[256][40];
  for (int c0 = 0; c0 < 256; c0 += 32) {
    for (int cc = 0; cc < 32; ++cc) {
      int c = c0 + cc;
      float v = xb[((size_t)c << 12) + n];
      tile[tid][cc] = (_Float16)((v - mean) * rstd * lnw[c] + lnb[c]);
    }
    __syncthreads();
#pragma unroll
    for (int rep = 0; rep < 4; ++rep) {
      int idx = rep * 2048 + tid * 8;
      int row = idx >> 5;
      int cs = idx & 31;
      *reinterpret_cast<int4*>(&hout[((size_t)(bn0 + row) << 8) + c0 + cs]) =
          *reinterpret_cast<const int4*>(&tile[row][cs]);
    }
    __syncthreads();
  }
}

// ---------------- QKVV GEMM: [65536,256] x [256,1024] -> f16 ----------------
// B supplied transposed: wqt[o][c].

__global__ __launch_bounds__(256) void k_gemm_qkvv(const _Float16* __restrict__ A,
                                                   const _Float16* __restrict__ Bt,
                                                   _Float16* __restrict__ Co) {
  const int tid = threadIdx.x;
  const int w = tid >> 5, lane = tid & 31;
  const int wm = w >> 1, wn = w & 1;
  const int m0 = blockIdx.x * 128;
  const int n0 = blockIdx.y * 64;
  __shared__ alignas(16) _Float16 As[128][40];
  __shared__ alignas(16) _Float16 Bst[64][40];  // [n][k]
  v8f acc[2][2] = {{vzero8(), vzero8()}, {vzero8(), vzero8()}};
  for (int kc = 0; kc < 256; kc += 32) {
    {
      int r = tid >> 1, sub = (tid & 1) * 16;
      const _Float16* src = A + (((size_t)(m0 + r)) << 8) + kc + sub;
      *reinterpret_cast<int4*>(&As[r][sub]) = *reinterpret_cast<const int4*>(src);
      *reinterpret_cast<int4*>(&As[r][sub + 8]) =
          *reinterpret_cast<const int4*>(src + 8);
    }
    {
      int r = tid >> 2, sub = (tid & 3) * 8;
      *reinterpret_cast<int4*>(&Bst[r][sub]) = *reinterpret_cast<const int4*>(
          Bt + (((size_t)(n0 + r)) << 8) + kc + sub);
    }
    __syncthreads();
    v16h a0 = ldsFrag(&As[wm * 32][0], 40);
    v16h a1 = ldsFrag(&As[wm * 32 + 16][0], 40);
    v16h b0 = ldsFrag(&Bst[wn * 32][0], 40);
    v16h b1 = ldsFrag(&Bst[wn * 32 + 16][0], 40);
    acc[0][0] = wmma_f16(a0, b0, acc[0][0]);
    acc[0][1] = wmma_f16(a0, b1, acc[0][1]);
    acc[1][0] = wmma_f16(a1, b0, acc[1][0]);
    acc[1][1] = wmma_f16(a1, b1, acc[1][1]);
    __syncthreads();
  }
  const int cn = lane & 15, hs = lane >> 4;
#pragma unroll
  for (int tt = 0; tt < 2; ++tt)
#pragma unroll
    for (int uu = 0; uu < 2; ++uu)
#pragma unroll
      for (int i = 0; i < 8; ++i) {
        int m = m0 + wm * 32 + tt * 16 + i + hs * 8;
        int nn = n0 + wn * 32 + uu * 16 + cn;
        Co[((size_t)m << 10) + nn] = (_Float16)acc[tt][uu][i];
      }
}

// ---------------- q/k row inverse norms over N ----------------

__global__ __launch_bounds__(256) void k_norms(const _Float16* __restrict__ qkvv,
                                               float* __restrict__ invnq,
                                               float* __restrict__ invnk) {
  const int tid = threadIdx.x;
  const int idx = blockIdx.x;  // 0..8191
  const int which = idx >> 12;
  const int r = idx & 4095;
  const int b = r >> 8;
  const int cc = r & 255;
  const _Float16* base = qkvv + (((size_t)b << 12) << 10) + which * 256 + cc;
  float s = 0.f;
  for (int n = tid; n < 4096; n += 256) {
    float v = (float)base[(size_t)n << 10];
    s += v * v;
  }
  __shared__ float red[256];
  red[tid] = s;
  __syncthreads();
  for (int st = 128; st > 0; st >>= 1) {
    if (tid < st) red[tid] += red[tid + st];
    __syncthreads();
  }
  if (tid == 0) {
    float inv = 1.f / fmaxf(sqrtf(red[0]), 1e-12f);
    (which ? invnk : invnq)[r] = inv;
  }
}

// ---------- channel-attention Gram (32x32 over N=4096) + softmax ----------
// Small kernel (128 blocks); keeps scalar transposed gathers.

__global__ __launch_bounds__(256) void k_gram(const _Float16* __restrict__ qkvv,
                                              const float* __restrict__ invnq,
                                              const float* __restrict__ invnk,
                                              _Float16* __restrict__ attnCa) {
  const int tid = threadIdx.x;
  const int w = tid >> 5, lane = tid & 31;
  const int bh = blockIdx.x;
  const int b = bh >> 3, hh = bh & 7;
  __shared__ alignas(16) _Float16 Aq[8][32][40];
  __shared__ alignas(16) _Float16 Ak[8][32][40];
  __shared__ float Gf[1024];
  for (int i = tid; i < 1024; i += 256) Gf[i] = 0.f;
  __syncthreads();
  v8f acc[2][2] = {{vzero8(), vzero8()}, {vzero8(), vzero8()}};
  const size_t rowbase = ((size_t)b << 12) << 10;
  const int qcol = hh * 32;
  for (int it = 0; it < 16; ++it) {
    int n = w * 512 + it * 32 + lane;
    const _Float16* src = qkvv + rowbase + ((size_t)n << 10) + qcol;
#pragma unroll
    for (int u = 0; u < 4; ++u)
      *reinterpret_cast<int4*>(&Aq[w][lane][u * 8]) =
          *reinterpret_cast<const int4*>(src + u * 8);
#pragma unroll
    for (int u = 0; u < 4; ++u)
      *reinterpret_cast<int4*>(&Ak[w][lane][u * 8]) =
          *reinterpret_cast<const int4*>(src + 256 + u * 8);
    __syncthreads();
    v16h a0 = ldsAT(&Aq[w][0][0], 40);
    v16h a1 = ldsAT(&Aq[w][0][16], 40);
    v16h b0 = ldsB(&Ak[w][0][0], 40);
    v16h b1 = ldsB(&Ak[w][0][16], 40);
    acc[0][0] = wmma_f16(a0, b0, acc[0][0]);
    acc[0][1] = wmma_f16(a0, b1, acc[0][1]);
    acc[1][0] = wmma_f16(a1, b0, acc[1][0]);
    acc[1][1] = wmma_f16(a1, b1, acc[1][1]);
    __syncthreads();
  }
  const int cn = lane & 15, hs = lane >> 4;
#pragma unroll
  for (int tt = 0; tt < 2; ++tt)
#pragma unroll
    for (int uu = 0; uu < 2; ++uu)
#pragma unroll
      for (int i = 0; i < 8; ++i) {
        int dd = tt * 16 + i + hs * 8;
        int ee = uu * 16 + cn;
        atomicAdd(&Gf[dd * 32 + ee], acc[tt][uu][i]);  // ds_add_f32
      }
  __syncthreads();
  if (tid < 32) {
    const int dd = tid;
    const float sq = invnq[bh * 32 + dd] * (1.f / 64.f);  // 1/sqrt(N)
    float g[32];
    float mx = -1e30f;
    for (int ee = 0; ee < 32; ++ee) {
      g[ee] = Gf[dd * 32 + ee] * sq * invnk[bh * 32 + ee];
      mx = fmaxf(mx, g[ee]);
    }
    float ssum = 0.f;
    for (int ee = 0; ee < 32; ++ee) {
      g[ee] = __expf(g[ee] - mx);
      ssum += g[ee];
    }
    const float rs = 1.f / ssum;
    // natural [dd][ee] layout == transposed-B layout for the CA apply GEMM
    for (int ee = 0; ee < 32; ++ee)
      attnCa[(size_t)bh * 1024 + dd * 32 + ee] = (_Float16)(g[ee] * rs);
  }
}

// ---- low-rank projections: A = ef^T (vectorized), B = k/v (scalar gather) ----
// Outputs in transposed-B layouts for the fused SA kernel:
//   kps[kk][dd] (scaled), vpt[dd][kk].

__global__ __launch_bounds__(256) void k_proj(const _Float16* __restrict__ qkvv,
                                              const _Float16* __restrict__ eft,
                                              const float* __restrict__ invnq,
                                              const float* __restrict__ temp2,
                                              _Float16* __restrict__ kps,
                                              _Float16* __restrict__ vpt) {
  const int tid = threadIdx.x;
  const int w = tid >> 5, lane = tid & 31;
  const int mt = w >> 1, nt = w & 1;
  const int bh = blockIdx.x;
  const int b = bh >> 3, hh = bh & 7;
  __shared__ alignas(16) _Float16 Ae[64][40];
  __shared__ alignas(16) _Float16 Bk[32][40];
  __shared__ alignas(16) _Float16 Bv[32][40];
  v8f acck = vzero8(), accv = vzero8();
  const size_t rowbase = ((size_t)b << 12) << 10;
  for (int n0 = 0; n0 < 4096; n0 += 32) {
    {
      int r = tid >> 2, sub = (tid & 3) * 8;
      *reinterpret_cast<int4*>(&Ae[r][sub]) = *reinterpret_cast<const int4*>(
          eft + ((size_t)r << 12) + n0 + sub);
    }
    if (tid < 128) {
      int r = tid >> 2, sub = (tid & 3) * 8;
      *reinterpret_cast<int4*>(&Bk[r][sub]) = *reinterpret_cast<const int4*>(
          qkvv + rowbase + ((size_t)(n0 + r) << 10) + 256 + hh * 32 + sub);
    } else {
      int t = tid - 128;
      int r = t >> 2, sub = (t & 3) * 8;
      *reinterpret_cast<int4*>(&Bv[r][sub]) = *reinterpret_cast<const int4*>(
          qkvv + rowbase + ((size_t)(n0 + r) << 10) + 768 + hh * 32 + sub);
    }
    __syncthreads();
    v16h a = ldsFrag(&Ae[mt * 16][0], 40);
    v16h bk = ldsB(&Bk[0][nt * 16], 40);
    v16h bv = ldsB(&Bv[0][nt * 16], 40);
    acck = wmma_f16(a, bk, acck);
    accv = wmma_f16(a, bv, accv);
    __syncthreads();
  }
  const int cn = lane & 15, hs = lane >> 4;
  const float tmp = temp2[hh];
#pragma unroll
  for (int i = 0; i < 8; ++i) {
    int kk = mt * 16 + i + hs * 8;
    int dd = nt * 16 + cn;
    kps[(size_t)bh * 2048 + kk * 32 + dd] =
        (_Float16)(acck[i] * invnq[bh * 32 + dd] * tmp);
    vpt[(size_t)bh * 2048 + dd * 64 + kk] = (_Float16)accv[i];
  }
}

// ---- fused spatial attention (logits->softmax->apply) + CA apply ----
// All fragments vectorized (A tiles natural, B tiles transposed).

__global__ __launch_bounds__(256) void k_sa_ca(const _Float16* __restrict__ qkvv,
                                               const _Float16* __restrict__ kps,
                                               const _Float16* __restrict__ vpt,
                                               const _Float16* __restrict__ attnCa,
                                               float* __restrict__ xca,
                                               float* __restrict__ xsa) {
  const int tid = threadIdx.x;
  const int w = tid >> 5, lane = tid & 31;
  const int bh = blockIdx.x;
  const int b = bh >> 3, hh = bh & 7;
  const int n0 = blockIdx.y * 64;
  __shared__ alignas(16) _Float16 Aq[64][40];    // q^T tile [n][dd]
  __shared__ alignas(16) _Float16 Av[64][40];    // v_ca^T tile [n][e]
  __shared__ alignas(16) _Float16 Bkpt[64][40];  // kps [kk][dd]  (Bt: [n][k])
  __shared__ alignas(16) _Float16 Bvpt[32][72];  // vpt [dd][kk]  (Bt: [n][k])
  __shared__ alignas(16) _Float16 BcaT[32][40];  // attn_ca [dd][e] (Bt)
  __shared__ float Lf[64][68];
  __shared__ alignas(16) _Float16 A2[64][72];
  const size_t rowbase = ((size_t)b << 12) << 10;
  {
    int r = tid >> 2, sub = (tid & 3) * 8;
    *reinterpret_cast<int4*>(&Aq[r][sub]) = *reinterpret_cast<const int4*>(
        qkvv + rowbase + ((size_t)(n0 + r) << 10) + hh * 32 + sub);
    *reinterpret_cast<int4*>(&Av[r][sub]) = *reinterpret_cast<const int4*>(
        qkvv + rowbase + ((size_t)(n0 + r) << 10) + 512 + hh * 32 + sub);
    *reinterpret_cast<int4*>(&Bkpt[r][sub]) = *reinterpret_cast<const int4*>(
        kps + (size_t)bh * 2048 + r * 32 + sub);
  }
  {
    int r = tid >> 3, c8 = (tid & 7) * 8;
    *reinterpret_cast<int4*>(&Bvpt[r][c8]) = *reinterpret_cast<const int4*>(
        vpt + (size_t)bh * 2048 + r * 64 + c8);
  }
  if (tid < 128) {
    int r = tid >> 2, sub = (tid & 3) * 8;
    *reinterpret_cast<int4*>(&BcaT[r][sub]) = *reinterpret_cast<const int4*>(
        attnCa + (size_t)bh * 1024 + r * 32 + sub);
  }
  __syncthreads();
  const int cn = lane & 15, hs = lane >> 4;
  {
    const int mt = w >> 1, ub = (w & 1) * 2;
    v16h a = ldsFrag(&Aq[mt * 16][0], 40);
#pragma unroll
    for (int uu = 0; uu < 2; ++uu) {
      int u = ub + uu;
      v16h bb = ldsFrag(&Bkpt[u * 16][0], 40);
      v8f acc = wmma_f16(a, bb, vzero8());
#pragma unroll
      for (int i = 0; i < 8; ++i)
        Lf[mt * 16 + i + hs * 8][u * 16 + cn] = acc[i];
    }
    const int u = w & 1;
    v16h av = ldsFrag(&Av[mt * 16][0], 40);
    v16h bb = ldsFrag(&BcaT[u * 16][0], 40);
    v8f acc = wmma_f16(av, bb, vzero8());
#pragma unroll
    for (int i = 0; i < 8; ++i) {
      int n = n0 + mt * 16 + i + hs * 8;
      int dd = u * 16 + cn;
      xca[(((size_t)bh << 12) + n) * 32 + dd] = acc[i];
    }
  }
  __syncthreads();
  if (tid < 64) {
    float mx = -1e30f;
    for (int kk2 = 0; kk2 < 64; ++kk2) mx = fmaxf(mx, Lf[tid][kk2]);
    float ssum = 0.f;
    for (int kk2 = 0; kk2 < 64; ++kk2) ssum += __expf(Lf[tid][kk2] - mx);
    const float rs = 1.f / ssum;
    for (int kk2 = 0; kk2 < 64; ++kk2)
      A2[tid][kk2] = (_Float16)(__expf(Lf[tid][kk2] - mx) * rs);
  }
  __syncthreads();
  {
    const int mt = w >> 1, u = w & 1;
    v16h a0 = ldsFrag(&A2[mt * 16][0], 72);
    v16h a1 = ldsFrag(&A2[mt * 16][32], 72);
    v16h b0 = ldsFrag(&Bvpt[u * 16][0], 72);
    v16h b1 = ldsFrag(&Bvpt[u * 16][32], 72);
    v8f acc = wmma_f16(a0, b0, vzero8());
    acc = wmma_f16(a1, b1, acc);
#pragma unroll
    for (int i = 0; i < 8; ++i) {
      int n = n0 + mt * 16 + i + hs * 8;
      int dd = u * 16 + cn;
      xsa[(((size_t)bh << 12) + n) * 32 + dd] = acc[i];
    }
  }
}

// ---- residual combine: attn = xs + gamma*(x_ca + x_sa), NHWC f32/f16 ----

__global__ __launch_bounds__(256) void k_combine(const float* __restrict__ x,
                                                 const float* __restrict__ gamma,
                                                 const float* __restrict__ xca,
                                                 const float* __restrict__ xsa,
                                                 float* __restrict__ askip,
                                                 _Float16* __restrict__ a16) {
  const int c = threadIdx.x;
  const int b = blockIdx.x;
  const int n0 = blockIdx.y * 64;
  const float g = gamma[c];
  const int hh = c >> 5, dd = c & 31;
  for (int i = 0; i < 64; ++i) {
    int n = n0 + i;
    int bn = (b << 12) + n;
    float xv = x[(((size_t)(b * 256 + c)) << 12) + n];
    float vca = xca[((((size_t)(b * 8 + hh)) << 12) + n) * 32 + dd];
    int f = (n << 8) + c;  // faithful x_sa permute/reshape remap
    int dd2 = f >> 15;
    int hh2 = (f >> 12) & 7;
    int nn = f & 4095;
    float vsa = xsa[((((size_t)(b * 8 + hh2)) << 12) + nn) * 32 + dd2];
    float a = xv + g * (vca + vsa);
    askip[((size_t)bn << 8) + c] = a;
    a16[((size_t)bn << 8) + c] = (_Float16)a;
  }
}

// ---- 3x3 conv as 9 shifted WMMA GEMM taps (bias cancels under train BN) ----

__global__ __launch_bounds__(256) void k_conv3(const _Float16* __restrict__ in16,
                                               const _Float16* __restrict__ w9t,
                                               float* __restrict__ outp) {
  const int tid = threadIdx.x;
  const int w = tid >> 5, lane = tid & 31;
  const int wm = w >> 1, wn = w & 1;
  const int m0 = blockIdx.x * 128;
  const int n0 = blockIdx.y * 64;
  __shared__ alignas(16) _Float16 As[128][40];
  __shared__ alignas(16) _Float16 Bst[64][40];  // [co][ci]
  v8f acc[2][2] = {{vzero8(), vzero8()}, {vzero8(), vzero8()}};
  const int r = tid >> 1, sub = (tid & 1) * 16;
  const int m = m0 + r;
  const int bb = m >> 12;
  const int hw = m & 4095;
  const int hh0 = hw >> 6, ww0 = hw & 63;
  const int rb = tid >> 2, subb = (tid & 3) * 8;
  for (int tap = 0; tap < 9; ++tap) {
    const int dy = tap / 3 - 1, dx = tap % 3 - 1;
    const int hp = hh0 + dy, wp = ww0 + dx;
    const bool valid = ((unsigned)hp < 64u) && ((unsigned)wp < 64u);
    const _Float16* arow = in16 + ((((size_t)bb << 12) + (hp << 6) + wp) << 8);
    for (int kc = 0; kc < 256; kc += 32) {
      if (valid) {
        const _Float16* src = arow + kc + sub;
        __builtin_prefetch(src + 32, 0, 1);
        *reinterpret_cast<int4*>(&As[r][sub]) =
            *reinterpret_cast<const int4*>(src);
        *reinterpret_cast<int4*>(&As[r][sub + 8]) =
            *reinterpret_cast<const int4*>(src + 8);
      } else {
        int4 z = {0, 0, 0, 0};
        *reinterpret_cast<int4*>(&As[r][sub]) = z;
        *reinterpret_cast<int4*>(&As[r][sub + 8]) = z;
      }
      *reinterpret_cast<int4*>(&Bst[rb][subb]) = *reinterpret_cast<const int4*>(
          w9t + (((size_t)tap * 256 + n0 + rb) << 8) + kc + subb);
      __syncthreads();
      v16h a0 = ldsFrag(&As[wm * 32][0], 40);
      v16h a1 = ldsFrag(&As[wm * 32 + 16][0], 40);
      v16h b0 = ldsFrag(&Bst[wn * 32][0], 40);
      v16h b1 = ldsFrag(&Bst[wn * 32 + 16][0], 40);
      acc[0][0] = wmma_f16(a0, b0, acc[0][0]);
      acc[0][1] = wmma_f16(a0, b1, acc[0][1]);
      acc[1][0] = wmma_f16(a1, b0, acc[1][0]);
      acc[1][1] = wmma_f16(a1, b1, acc[1][1]);
      __syncthreads();
    }
  }
  const int cn = lane & 15, hs = lane >> 4;
#pragma unroll
  for (int tt = 0; tt < 2; ++tt)
#pragma unroll
    for (int uu = 0; uu < 2; ++uu)
#pragma unroll
      for (int i = 0; i < 8; ++i) {
        int mm = m0 + wm * 32 + tt * 16 + i + hs * 8;
        int nn = n0 + wn * 32 + uu * 16 + cn;
        outp[((size_t)mm << 8) + nn] = acc[tt][uu][i];
      }
}

// ---- training-mode batchnorm stats + fused normalize/activation ----

__global__ __launch_bounds__(256) void k_bn_stats(const float* __restrict__ v,
                                                  float* __restrict__ mean,
                                                  float* __restrict__ rstd) {
  const int c = blockIdx.x;
  const int tid = threadIdx.x;
  float s = 0.f, sq = 0.f;
  for (int m = tid; m < 65536; m += 256) {
    float t = v[((size_t)m << 8) + c];
    s += t;
    sq += t * t;
  }
  __shared__ float r1[256], r2[256];
  r1[tid] = s;
  r2[tid] = sq;
  __syncthreads();
  for (int st = 128; st > 0; st >>= 1) {
    if (tid < st) {
      r1[tid] += r1[tid + st];
      r2[tid] += r2[tid + st];
    }
    __syncthreads();
  }
  if (tid == 0) {
    float mu = r1[0] * (1.f / 65536.f);
    float var = r2[0] * (1.f / 65536.f) - mu * mu;
    mean[c] = mu;
    rstd[c] = rsqrtf(var + 1e-5f);
  }
}

__global__ __launch_bounds__(256) void k_bn_act(const float* __restrict__ v,
                                                const float* __restrict__ mean,
                                                const float* __restrict__ rstd,
                                                const float* __restrict__ bw,
                                                const float* __restrict__ bb,
                                                const float* __restrict__ skip,
                                                _Float16* __restrict__ out16,
                                                int mode) {
  size_t idx = (size_t)blockIdx.x * 256 + threadIdx.x;
  int c = (int)(idx & 255);
  float t = (v[idx] - mean[c]) * rstd[c] * bw[c] + bb[c];
  if (mode == 1) t += skip[idx];
  float y = (t >= 0.f) ? t : 0.01f * t;
  out16[idx] = (_Float16)y;
}

// ---- final 1x1 conv GEMM + skip, stored transposed to NCHW ----
// B supplied transposed: w8t[co][ci].

__global__ __launch_bounds__(256) void k_final(const _Float16* __restrict__ A,
                                               const _Float16* __restrict__ Bt,
                                               const float* __restrict__ askip,
                                               const float* __restrict__ bias,
                                               float* __restrict__ outp) {
  const int tid = threadIdx.x;
  const int w = tid >> 5, lane = tid & 31;
  const int wm = w >> 1, wn = w & 1;
  const int m0 = blockIdx.x * 128;
  const int n0 = blockIdx.y * 64;
  __shared__ alignas(16) _Float16 As[128][40];
  __shared__ alignas(16) _Float16 Bst[64][40];
  v8f acc[2][2] = {{vzero8(), vzero8()}, {vzero8(), vzero8()}};
  for (int kc = 0; kc < 256; kc += 32) {
    {
      int r = tid >> 1, sub = (tid & 1) * 16;
      const _Float16* src = A + (((size_t)(m0 + r)) << 8) + kc + sub;
      *reinterpret_cast<int4*>(&As[r][sub]) = *reinterpret_cast<const int4*>(src);
      *reinterpret_cast<int4*>(&As[r][sub + 8]) =
          *reinterpret_cast<const int4*>(src + 8);
    }
    {
      int r = tid >> 2, sub = (tid & 3) * 8;
      *reinterpret_cast<int4*>(&Bst[r][sub]) = *reinterpret_cast<const int4*>(
          Bt + (((size_t)(n0 + r)) << 8) + kc + sub);
    }
    __syncthreads();
    v16h a0 = ldsFrag(&As[wm * 32][0], 40);
    v16h a1 = ldsFrag(&As[wm * 32 + 16][0], 40);
    v16h b0 = ldsFrag(&Bst[wn * 32][0], 40);
    v16h b1 = ldsFrag(&Bst[wn * 32 + 16][0], 40);
    acc[0][0] = wmma_f16(a0, b0, acc[0][0]);
    acc[0][1] = wmma_f16(a0, b1, acc[0][1]);
    acc[1][0] = wmma_f16(a1, b0, acc[1][0]);
    acc[1][1] = wmma_f16(a1, b1, acc[1][1]);
    __syncthreads();
  }
  const int cn = lane & 15, hs = lane >> 4;
#pragma unroll
  for (int tt = 0; tt < 2; ++tt)
#pragma unroll
    for (int uu = 0; uu < 2; ++uu)
#pragma unroll
      for (int i = 0; i < 8; ++i) {
        int mm = m0 + wm * 32 + tt * 16 + i + hs * 8;
        int co = n0 + wn * 32 + uu * 16 + cn;
        float val = acc[tt][uu][i] + bias[co] + askip[((size_t)mm << 8) + co];
        int b = mm >> 12, n = mm & 4095;
        outp[(((size_t)(b * 256 + co)) << 12) + n] = val;  // NCHW
      }
}

// ---------------------------------------------------------------------------

extern "C" void kernel_launch(void* const* d_in, const int* in_sizes, int n_in,
                              void* d_out, int out_size, void* d_ws, size_t ws_size,
                              hipStream_t stream) {
  (void)in_sizes; (void)n_in; (void)out_size; (void)ws_size;
  const float* x       = (const float*)d_in[0];
  const float* w_qkvv  = (const float*)d_in[1];
  const float* ef      = (const float*)d_in[2];
  const float* temp2   = (const float*)d_in[3];
  const float* ln_w    = (const float*)d_in[4];
  const float* ln_b    = (const float*)d_in[5];
  const float* gamma   = (const float*)d_in[6];
  const float* conv1_w = (const float*)d_in[7];
  // d_in[8] (conv1_b) unused: cancelled by training-mode batchnorm
  const float* bn1_w   = (const float*)d_in[9];
  const float* bn1_b   = (const float*)d_in[10];
  const float* conv2_w = (const float*)d_in[11];
  // d_in[12] (conv2_b) unused: cancelled by training-mode batchnorm
  const float* bn2_w   = (const float*)d_in[13];
  const float* bn2_b   = (const float*)d_in[14];
  const float* conv8_w = (const float*)d_in[15];
  const float* conv8_b = (const float*)d_in[16];
  float* outp = (float*)d_out;

  const size_t BN = 65536;  // B*H*W
  char* ws = (char*)d_ws;
  size_t off = 0;
  auto alloc = [&](size_t bytes) -> char* {
    off = (off + 255) & ~(size_t)255;
    char* p = ws + off;
    off += bytes;
    return p;
  };
  _Float16* hbuf   = (_Float16*)alloc(BN * 256 * 2);   // 32 MiB
  _Float16* qkvv16 = (_Float16*)alloc(BN * 1024 * 2);  // 128 MiB
  float*    xca    = (float*)alloc(BN * 256 * 4);      // 64 MiB
  float*    xsa    = (float*)alloc(BN * 256 * 4);      // 64 MiB
  _Float16* wqt    = (_Float16*)alloc((size_t)1024 * 256 * 2);
  _Float16* eft    = (_Float16*)alloc((size_t)64 * 4096 * 2);
  float*    invnq  = (float*)alloc(4096 * 4);
  float*    invnk  = (float*)alloc(4096 * 4);
  _Float16* attnCa = (_Float16*)alloc((size_t)128 * 1024 * 2);
  _Float16* kps    = (_Float16*)alloc((size_t)128 * 2048 * 2);
  _Float16* vpt    = (_Float16*)alloc((size_t)128 * 2048 * 2);
  _Float16* w9a    = (_Float16*)alloc((size_t)9 * 256 * 256 * 2);
  _Float16* w9b    = (_Float16*)alloc((size_t)9 * 256 * 256 * 2);
  _Float16* w8t    = (_Float16*)alloc((size_t)256 * 256 * 2);
  float* bnm1 = (float*)alloc(1024);
  float* bnr1 = (float*)alloc(1024);
  float* bnm2 = (float*)alloc(1024);
  float* bnr2 = (float*)alloc(1024);
  // aliases with disjoint lifetimes:
  _Float16* a16   = (_Float16*)qkvv16;
  float*    askip = (float*)((char*)qkvv16 + BN * 256 * 2);
  float*    ctmp  = xca;
  _Float16* y16   = hbuf;
  _Float16* z16   = (_Float16*)xsa;

  k_cvt_wq<<<1024, 256, 0, stream>>>(w_qkvv, wqt);
  k_cvt_ef<<<dim3(64, 16), 256, 0, stream>>>(ef, eft);
  k_cvt_w9<<<dim3(9, 256), 256, 0, stream>>>(conv1_w, w9a);
  k_cvt_w9<<<dim3(9, 256), 256, 0, stream>>>(conv2_w, w9b);
  k_cvt_w8<<<256, 256, 0, stream>>>(conv8_w, w8t);
  k_layernorm<<<256, 256, 0, stream>>>(x, ln_w, ln_b, hbuf);
  k_gemm_qkvv<<<dim3(512, 16), 256, 0, stream>>>(hbuf, wqt, qkvv16);
  k_norms<<<8192, 256, 0, stream>>>(qkvv16, invnq, invnk);
  k_gram<<<128, 256, 0, stream>>>(qkvv16, invnq, invnk, attnCa);
  k_proj<<<128, 256, 0, stream>>>(qkvv16, eft, invnq, temp2, kps, vpt);
  k_sa_ca<<<dim3(128, 64), 256, 0, stream>>>(qkvv16, kps, vpt, attnCa, xca, xsa);
  k_combine<<<dim3(16, 64), 256, 0, stream>>>(x, gamma, xca, xsa, askip, a16);
  k_conv3<<<dim3(512, 4), 256, 0, stream>>>(a16, w9a, ctmp);
  k_bn_stats<<<256, 256, 0, stream>>>(ctmp, bnm1, bnr1);
  k_bn_act<<<65536, 256, 0, stream>>>(ctmp, bnm1, bnr1, bn1_w, bn1_b, nullptr, y16, 0);
  k_conv3<<<dim3(512, 4), 256, 0, stream>>>(y16, w9b, ctmp);
  k_bn_stats<<<256, 256, 0, stream>>>(ctmp, bnm2, bnr2);
  k_bn_act<<<65536, 256, 0, stream>>>(ctmp, bnm2, bnr2, bn2_w, bn2_b, askip, z16, 1);
  k_final<<<dim3(512, 4), 256, 0, stream>>>(z16, w8t, askip, conv8_b, outp);
}
